// GNN_3633542333061
// MI455X (gfx1250) — matrix operand order repaired and checked by reading, not compiled, not run on hardware
//
#include <hip/hip_runtime.h>
#include <hip/hip_bf16.h>

#define BSZ   2
#define NN    50000
#define NEDGE 250000

typedef __attribute__((ext_vector_type(16))) __bf16 v16bf;
typedef __attribute__((ext_vector_type(8)))  __bf16 v8bf;
typedef __attribute__((ext_vector_type(8)))  float  v8f;

static __device__ __forceinline__ __bf16 f2bf(float f) { return (__bf16)f; }
static __device__ __forceinline__ unsigned short f2bf_u(float f) {
  return __builtin_bit_cast(unsigned short, (__bf16)f);
}

static __device__ __forceinline__ v8f wmma_bf16(v16bf a, v16bf b, v8f c) {
  return __builtin_amdgcn_wmma_f32_16x16x32_bf16(false, a, false, b, (short)0, c, false, false);
}

// ---------------------------------------------------------------------------
// Pack f32 weight W[K][128] into WMMA B-fragment order (bf16):
// P[((kb*8+nb)*32 + lane)*16 + j] = bf16(W[kb*32 + (lane<16?0:8) + (j<8 ? j : j+8)][nb*16 + lane%16])
// ---------------------------------------------------------------------------
__global__ void pack_weights(const float* __restrict__ W, unsigned short* __restrict__ P, int total) {
  int tid = blockIdx.x * blockDim.x + threadIdx.x;
  if (tid >= total) return;
  int j    = tid & 15;
  int lane = (tid >> 4) & 31;
  int t    = tid >> 9;
  int nb   = t & 7;
  int kb   = t >> 3;
  int k    = kb * 32 + ((lane >> 4) << 3) + (j < 8 ? j : j + 8);
  int col  = nb * 16 + (lane & 15);
  P[tid] = f2bf_u(W[k * 128 + col]);
}

// A-fragment load from ONE buffer: p already points at this lane's feature f0.
// Runs [0,8) and [16,24): the A-layout for 16-bit 16x32 fragments.
static __device__ __forceinline__ v16bf loadA8(const float* p) {
  float4 q0 = *(const float4*)(p);
  float4 q1 = *(const float4*)(p + 4);
  float4 q2 = *(const float4*)(p + 16);
  float4 q3 = *(const float4*)(p + 20);
  v16bf a;
  a[0]  = f2bf(q0.x); a[1]  = f2bf(q0.y); a[2]  = f2bf(q0.z); a[3]  = f2bf(q0.w);
  a[4]  = f2bf(q1.x); a[5]  = f2bf(q1.y); a[6]  = f2bf(q1.z); a[7]  = f2bf(q1.w);
  a[8]  = f2bf(q2.x); a[9]  = f2bf(q2.y); a[10] = f2bf(q2.z); a[11] = f2bf(q2.w);
  a[12] = f2bf(q3.x); a[13] = f2bf(q3.y); a[14] = f2bf(q3.z); a[15] = f2bf(q3.w);
  return a;
}

static __device__ __forceinline__ v16bf loadA8s(const float* p, float sc) {
  float4 q0 = *(const float4*)(p);
  float4 q1 = *(const float4*)(p + 4);
  float4 q2 = *(const float4*)(p + 16);
  float4 q3 = *(const float4*)(p + 20);
  v16bf a;
  a[0]  = f2bf(q0.x * sc); a[1]  = f2bf(q0.y * sc); a[2]  = f2bf(q0.z * sc); a[3]  = f2bf(q0.w * sc);
  a[4]  = f2bf(q1.x * sc); a[5]  = f2bf(q1.y * sc); a[6]  = f2bf(q1.z * sc); a[7]  = f2bf(q1.w * sc);
  a[8]  = f2bf(q2.x * sc); a[9]  = f2bf(q2.y * sc); a[10] = f2bf(q2.z * sc); a[11] = f2bf(q2.w * sc);
  a[12] = f2bf(q3.x * sc); a[13] = f2bf(q3.y * sc); a[14] = f2bf(q3.z * sc); a[15] = f2bf(q3.w * sc);
  return a;
}

static __device__ __forceinline__ v16bf ldsA(const unsigned short* rowp, int f0) {
  v8bf lo = *(const v8bf*)(rowp + f0);
  v8bf hi = *(const v8bf*)(rowp + f0 + 16);
  v16bf a;
#pragma unroll
  for (int j = 0; j < 8; ++j) { a[j] = lo[j]; a[j + 8] = hi[j]; }
  return a;
}

// ---------------------------------------------------------------------------
// Edge MLP + scatter-add. One wave per block, 32 edges per wave
// (two 16-row WMMA tiles sharing every weight fragment).
// ---------------------------------------------------------------------------
__global__ __launch_bounds__(32) void edge_mlp_kernel(
    const float* __restrict__ V, const float* __restrict__ E,
    const int* __restrict__ edges,
    const unsigned short* __restrict__ W1, const unsigned short* __restrict__ W2,
    const float* __restrict__ B1, const float* __restrict__ B2,
    float* __restrict__ outE,
    float* __restrict__ sum0, float* __restrict__ sum1,
    unsigned int* __restrict__ cnt0, unsigned int* __restrict__ cnt1)
{
  const int lane  = threadIdx.x;
  const int tilesPerB = (NEDGE + 31) / 32;           // 7813
  const int b     = blockIdx.x / tilesPerB;
  const int t     = blockIdx.x % tilesPerB;
  const int colb  = lane & 15;                        // C column / A row
  const int rbase = (lane >> 4) << 3;                 // C row base (0 or 8)
  const int klo   = (lane >> 4) << 3;                 // A K base (0 or 8)
  const int row0  = t * 32;                           // uniform tile base

  bool okU[2];
  int snd[2], rcv[2];
  const float *vs[2], *vr[2], *er[2];
#pragma unroll
  for (int u = 0; u < 2; ++u) {
    okU[u] = (row0 + u * 16) < NEDGE;                 // sub-tile all-or-nothing
    int rc = okU[u] ? (row0 + u * 16 + colb) : 0;
    long long eb = ((long long)b * NEDGE + rc);
    snd[u] = edges[eb * 2 + 0];
    rcv[u] = edges[eb * 2 + 1];
    vs[u]  = V + ((long long)b * NN + snd[u]) * 128;
    vr[u]  = V + ((long long)b * NN + rcv[u]) * 128;
    er[u]  = E + eb * 128;
    if (okU[u] && lane < 16) {
      atomicAdd(&cnt0[b * NN + snd[u]], 1u);
      atomicAdd(&cnt1[b * NN + rcv[u]], 1u);
    }
  }

  // ---- Layer 1: [32 x 384] x [384 x 128]; region resolved at compile time ----
  v8f acc[2][8] = {};
#pragma unroll
  for (int kb = 0; kb < 12; ++kb) {
    const int fb = (kb < 4) ? kb * 32 : (kb < 8) ? (kb - 4) * 32 : (kb - 8) * 32;
    const float* p0 = ((kb < 4) ? vs[0] : (kb < 8) ? vr[0] : er[0]) + fb + klo;
    const float* p1 = ((kb < 4) ? vs[1] : (kb < 8) ? vr[1] : er[1]) + fb + klo;
    v16bf a0 = loadA8(p0);
    v16bf a1 = loadA8(p1);
    const unsigned short* wp = W1 + kb * 4096 + lane * 16;
#pragma unroll
    for (int nb = 0; nb < 8; ++nb) {
      v16bf bb = *(const v16bf*)(wp + nb * 512);
      acc[0][nb] = wmma_bf16(a0, bb, acc[0][nb]);
      acc[1][nb] = wmma_bf16(a1, bb, acc[1][nb]);
    }
  }

  // ---- bias + SiLU, stage to LDS (C layout -> A layout re-map) ----
  __shared__ __align__(16) unsigned short hbuf[32 * 128];
#pragma unroll
  for (int u = 0; u < 2; ++u)
#pragma unroll
    for (int nb = 0; nb < 8; ++nb) {
      float bias = B1[nb * 16 + colb];
#pragma unroll
      for (int r = 0; r < 8; ++r) {
        float x = acc[u][nb][r] + bias;
        x = x / (1.0f + __expf(-x));
        hbuf[(u * 16 + rbase + r) * 128 + nb * 16 + colb] = f2bf_u(x);
      }
    }
  __syncthreads();

  // ---- Layer 2: [32 x 128] x [128 x 128] ----
  v8f acc2[2][8] = {};
#pragma unroll
  for (int kb = 0; kb < 4; ++kb) {
    int f0 = kb * 32 + klo;
    v16bf a0 = ldsA(&hbuf[colb * 128], f0);
    v16bf a1 = ldsA(&hbuf[(16 + colb) * 128], f0);
    const unsigned short* wp = W2 + kb * 4096 + lane * 16;
#pragma unroll
    for (int nb = 0; nb < 8; ++nb) {
      v16bf bb = *(const v16bf*)(wp + nb * 512);
      acc2[0][nb] = wmma_bf16(a0, bb, acc2[0][nb]);
      acc2[1][nb] = wmma_bf16(a1, bb, acc2[1][nb]);
    }
  }

  // ---- store edge embeddings + scatter-add halves ----
  float bias2[8];
#pragma unroll
  for (int nb = 0; nb < 8; ++nb) bias2[nb] = B2[nb * 16 + colb];

#pragma unroll
  for (int u = 0; u < 2; ++u) {
    if (!okU[u]) continue;                            // uniform branch
    float* oe = outE + ((long long)b * NEDGE + row0 + u * 16) * 128 + colb;
#pragma unroll
    for (int r = 0; r < 8; ++r) {
      int m = rbase + r;
      int sm = __shfl(snd[u], m, 32);
      int rm = __shfl(rcv[u], m, 32);
      float* orow = oe + m * 128;
      float* s0p  = sum0 + ((long long)b * NN + sm) * 64 + colb;
      float* s1p  = sum1 + ((long long)b * NN + rm) * 64 + colb;
#pragma unroll
      for (int nb = 0; nb < 8; ++nb) {
        float val = acc2[u][nb][r] + bias2[nb];
        orow[nb * 16] = val;
        if (nb < 4) atomicAdd(s0p + nb * 16, val);
        else        atomicAdd(s1p + (nb - 4) * 16, val);
      }
    }
  }
}

// ---------------------------------------------------------------------------
// Node MLP. One wave per block, 16 nodes per wave.
// Input = [V(128) | sum0/cnt0 (64) | sum1/cnt1 (64)] = 256 features.
// ---------------------------------------------------------------------------
__global__ __launch_bounds__(32) void node_mlp_kernel(
    const float* __restrict__ V,
    const float* __restrict__ sum0, const float* __restrict__ sum1,
    const unsigned int* __restrict__ cnt0, const unsigned int* __restrict__ cnt1,
    const unsigned short* __restrict__ W1, const unsigned short* __restrict__ W2,
    const float* __restrict__ B1, const float* __restrict__ B2,
    float* __restrict__ outN)
{
  const int lane  = threadIdx.x;
  const int tilesPerB = NN / 16;                     // 3125 (exact)
  const int b     = blockIdx.x / tilesPerB;
  const int t     = blockIdx.x % tilesPerB;
  const int colb  = lane & 15;
  const int rbase = (lane >> 4) << 3;
  const int klo   = (lane >> 4) << 3;

  int row = t * 16 + colb;
  long long base = (long long)b * NN + row;
  float i0 = 1.0f / fmaxf((float)cnt0[base], 1.0f);
  float i1 = 1.0f / fmaxf((float)cnt1[base], 1.0f);
  const float* vp = V    + base * 128;
  const float* s0 = sum0 + base * 64;
  const float* s1 = sum1 + base * 64;

  // ---- Layer 1: [16 x 256] x [256 x 128]; region static per kb ----
  v8f acc[8] = {};
#pragma unroll
  for (int kb = 0; kb < 8; ++kb) {
    v16bf a;
    if (kb < 4)      a = loadA8 (vp + kb * 32 + klo);
    else if (kb < 6) a = loadA8s(s0 + (kb - 4) * 32 + klo, i0);
    else             a = loadA8s(s1 + (kb - 6) * 32 + klo, i1);
    const unsigned short* wp = W1 + kb * 4096 + lane * 16;
#pragma unroll
    for (int nb = 0; nb < 8; ++nb) {
      v16bf bb = *(const v16bf*)(wp + nb * 512);
      acc[nb] = wmma_bf16(a, bb, acc[nb]);
    }
  }

  __shared__ __align__(16) unsigned short hbuf[16 * 128];
#pragma unroll
  for (int nb = 0; nb < 8; ++nb) {
    float bias = B1[nb * 16 + colb];
#pragma unroll
    for (int r = 0; r < 8; ++r) {
      float x = acc[nb][r] + bias;
      x = x / (1.0f + __expf(-x));
      hbuf[(rbase + r) * 128 + nb * 16 + colb] = f2bf_u(x);
    }
  }
  __syncthreads();

  // ---- Layer 2: [16 x 128] x [128 x 128] ----
  v8f acc2[8] = {};
#pragma unroll
  for (int kb = 0; kb < 4; ++kb) {
    int f0 = kb * 32 + klo;
    v16bf a = ldsA(&hbuf[colb * 128], f0);
    const unsigned short* wp = W2 + kb * 4096 + lane * 16;
#pragma unroll
    for (int nb = 0; nb < 8; ++nb) {
      v16bf bb = *(const v16bf*)(wp + nb * 512);
      acc2[nb] = wmma_bf16(a, bb, acc2[nb]);
    }
  }

  float* on = outN + ((long long)b * NN + t * 16) * 128 + colb;
#pragma unroll
  for (int nb = 0; nb < 8; ++nb) {
    float bias = B2[nb * 16 + colb];
#pragma unroll
    for (int r = 0; r < 8; ++r) {
      on[(rbase + r) * 128 + nb * 16] = acc2[nb][r] + bias;
    }
  }
}

// ---------------------------------------------------------------------------
extern "C" void kernel_launch(void* const* d_in, const int* in_sizes, int n_in,
                              void* d_out, int out_size, void* d_ws, size_t ws_size,
                              hipStream_t stream) {
  const float* V     = (const float*)d_in[0];
  const float* E     = (const float*)d_in[1];
  const int*   edges = (const int*)d_in[2];
  const float* fe_w1 = (const float*)d_in[3];
  const float* fe_b1 = (const float*)d_in[4];
  const float* fe_w2 = (const float*)d_in[5];
  const float* fe_b2 = (const float*)d_in[6];
  const float* fn_w1 = (const float*)d_in[7];
  const float* fn_b1 = (const float*)d_in[8];
  const float* fn_w2 = (const float*)d_in[9];
  const float* fn_b2 = (const float*)d_in[10];

  // Workspace layout:
  //   [0, 229376)          packed bf16 weights (w1e 384x128, w2e 128x128, w1n 256x128, w2n 128x128)
  //   [229376, +25.6MB)    sum0  [BS, N, 64] f32
  //   [..,    +25.6MB)     sum1  [BS, N, 64] f32
  //   [..,    +400KB)      cnt0  [BS, N] u32
  //   [..,    +400KB)      cnt1  [BS, N] u32
  unsigned short* pw1e = (unsigned short*)d_ws;
  unsigned short* pw2e = pw1e + 384 * 128;
  unsigned short* pw1n = pw2e + 128 * 128;
  unsigned short* pw2n = pw1n + 256 * 128;
  char* ws2 = (char*)d_ws + 229376;
  float* sum0 = (float*)ws2;
  float* sum1 = sum0 + (size_t)BSZ * NN * 64;
  unsigned int* cnt0 = (unsigned int*)(sum1 + (size_t)BSZ * NN * 64);
  unsigned int* cnt1 = cnt0 + (size_t)BSZ * NN;

  size_t zero_bytes = (size_t)BSZ * NN * 64 * 2 * sizeof(float)
                    + (size_t)BSZ * NN * 2 * sizeof(unsigned int);
  hipMemsetAsync(ws2, 0, zero_bytes, stream);

  pack_weights<<<(384 * 128 + 255) / 256, 256, 0, stream>>>(fe_w1, pw1e, 384 * 128);
  pack_weights<<<(128 * 128 + 255) / 256, 256, 0, stream>>>(fe_w2, pw2e, 128 * 128);
  pack_weights<<<(256 * 128 + 255) / 256, 256, 0, stream>>>(fn_w1, pw1n, 256 * 128);
  pack_weights<<<(128 * 128 + 255) / 256, 256, 0, stream>>>(fn_w2, pw2n, 128 * 128);

  float* outN = (float*)d_out;
  float* outE = outN + (size_t)BSZ * NN * 128;

  edge_mlp_kernel<<<BSZ * ((NEDGE + 31) / 32), 32, 0, stream>>>(
      V, E, edges, pw1e, pw2e, fe_b1, fe_b2, outE, sum0, sum1, cnt0, cnt1);
  node_mlp_kernel<<<BSZ * (NN / 16), 32, 0, stream>>>(
      V, sum0, sum1, cnt0, cnt1, pw1n, pw2n, fn_b1, fn_b2, outN);
}
